// VNMRI_36790689857866
// MI455X (gfx1250) — matrix operand order repaired and checked by reading, not compiled, not run on hardware
//
#include <hip/hip_runtime.h>
#include <math.h>

// VN-MRI recon for MI455X (gfx1250, wave32, WMMA).
// Convs = implicit GEMM on v_wmma_f32_16x16x32_f16 with fragment-layout LDS:
// A/B operands are fetched as aligned ds_load_b128 pairs (16 contiguous
// K-halves per lane), K re-indexed as (plane,row,vp16) so per-chunk im2col
// staging is a contiguous copy (no integer division in the hot loop).
// Patch cores stage via GLOBAL_LOAD_ASYNC_TO_LDS_B128 from f16 mirrors.

typedef _Float16 half_t;
typedef __attribute__((ext_vector_type(8)))  _Float16 v8h;
typedef __attribute__((ext_vector_type(16))) _Float16 v16h;
typedef __attribute__((ext_vector_type(8)))  float    v8f;
typedef __attribute__((ext_vector_type(4)))  int      v4i;

#define TT   6
#define FF   24
#define HH   256
#define WW   256
#define HW2  65536
#define NPIX 393216      // T*H*W
#define KSZ  11
#define TAPS 121
#define KN   35

// -------- async global->LDS (16B) with safe fallback ------------------------
#if defined(__has_builtin)
#if __has_builtin(__builtin_amdgcn_global_load_async_to_lds_b128)
#define ASYNC_LDS 1
#endif
#endif

typedef __attribute__((address_space(1))) v4i* gv4i_p;
typedef __attribute__((address_space(3))) v4i* lv4i_p;

__device__ __forceinline__ void copy16_g2l(const half_t* g, half_t* l) {
#ifdef ASYNC_LDS
  __builtin_amdgcn_global_load_async_to_lds_b128((gv4i_p)g, (lv4i_p)l, 0, 0);
#else
  *(v8h*)l = *(const v8h*)g;
#endif
}
__device__ __forceinline__ void wait_g2l() {
#ifdef ASYNC_LDS
  asm volatile("s_wait_asynccnt 0x0" ::: "memory");
#endif
}

// ---------------------------------------------------------------- reductions
__global__ __launch_bounds__(256) void vn_reduce(const float* __restrict__ xr,
                                                 const float* __restrict__ xi,
                                                 const float* __restrict__ mask,
                                                 float* __restrict__ pnum,
                                                 float* __restrict__ pden) {
  __shared__ float s1[256], s2[256];
  int tid = threadIdx.x;
  int gid = blockIdx.x * 256 + tid;
  float a = 0.f, b = 0.f;
  for (int i = gid; i < NPIX; i += 256 * 256)
    a += xr[i] * xr[i] + xi[i] * xi[i];
  if (gid < TT * WW) b = mask[gid];
  s1[tid] = a; s2[tid] = b;
  __syncthreads();
  for (int off = 128; off > 0; off >>= 1) {
    if (tid < off) { s1[tid] += s1[tid + off]; s2[tid] += s2[tid + off]; }
    __syncthreads();
  }
  if (tid == 0) { pnum[blockIdx.x] = s1[0]; pden[blockIdx.x] = s2[0]; }
}

__global__ void vn_finalize(const float* __restrict__ pnum,
                            const float* __restrict__ pden,
                            float* __restrict__ scal) {
  float n = 0.f, d = 0.f;
  for (int i = 0; i < 256; ++i) { n += pnum[i]; d += pden[i]; }
  scal[0] = n; scal[1] = d;
  scal[2] = sqrtf(n / (d > 0.f ? d : 1.f));
}

// ------------------------------------------------------------ pointwise prep
__global__ __launch_bounds__(256) void vn_prep(const float* __restrict__ xr,
                                               const float* __restrict__ xi,
                                               const float* __restrict__ scal,
                                               float* __restrict__ ksr, float* __restrict__ ksi,
                                               float* __restrict__ tr,  float* __restrict__ ti,
                                               float* __restrict__ mr_, float* __restrict__ mi_) {
  int i = blockIdx.x * 256 + threadIdx.x;
  if (i >= NPIX) return;
  float inv = 1.f / scal[2];
  float a = xr[i] * inv, b = xi[i] * inv;
  ksr[i] = a; ksi[i] = b;
  tr[i] = a;  ti[i] = b;
  mr_[i] = 0.f; mi_[i] = 0.f;
}

// ------------------------------------------------- centered ortho 256-pt FFT
__global__ __launch_bounds__(256) void vn_fft256(const float* __restrict__ sr,
                                                 const float* __restrict__ si,
                                                 float* __restrict__ dr,
                                                 float* __restrict__ di,
                                                 int mode, float sign) {
  __shared__ float lr[256], li[256];
  int n = threadIdx.x;
  int b = blockIdx.x;
  int base, stride;
  if (mode == 0) { base = b * 256; stride = 1; }
  else           { base = (b >> 8) * HW2 + (b & 255); stride = 256; }
  int rev = __brev((unsigned)n) >> 24;
  int ns  = (rev + 128) & 255;                 // ifftshift on input
  lr[n] = sr[base + ns * stride];
  li[n] = si[base + ns * stride];
  __syncthreads();
  for (int s = 0; s < 8; ++s) {
    int hstep = 1 << s, mlen = hstep << 1;
    if (n < 128) {
      int j = n & (hstep - 1);
      int p = ((n >> s) << (s + 1)) + j;
      int q = p + hstep;
      float ang = sign * 6.283185307179586f * (float)j / (float)mlen;
      float sn, cs;
      __sincosf(ang, &sn, &cs);
      float xr_ = lr[q], xi_ = li[q];
      float tr_ = xr_ * cs - xi_ * sn;
      float ti_ = xr_ * sn + xi_ * cs;
      float ur = lr[p], ui = li[p];
      lr[p] = ur + tr_; li[p] = ui + ti_;
      lr[q] = ur - tr_; li[q] = ui - ti_;
    }
    __syncthreads();
  }
  int kd = (n + 128) & 255;                    // fftshift on output
  dr[base + kd * stride] = lr[n] * 0.0625f;
  di[base + kd * stride] = li[n] * 0.0625f;
}

__global__ __launch_bounds__(256) void vn_scale_k0(const float* __restrict__ tr,
                                                   const float* __restrict__ ti,
                                                   const float* __restrict__ k0,
                                                   float* __restrict__ Xr,
                                                   float* __restrict__ Xi,
                                                   half_t* __restrict__ Xhr,
                                                   half_t* __restrict__ Xhi) {
  int i = blockIdx.x * 256 + threadIdx.x;
  if (i >= NPIX) return;
  float k = k0[0];
  float a = tr[i] * k, b = ti[i] * k;
  Xr[i] = a; Xi[i] = b;
  Xhr[i] = (half_t)a; Xhi[i] = (half_t)b;
}

// ------------------------------------------------ forward complex conv: WMMA
// K order: k=(c*11+u)*16+vp, vp padded 11->16 (B zero on pads). KPAD=352=11 ch.
// M=64 px/WG (4 waves). A/B fragments: 2x ds_load_b128 each.
#define FKP 352
__global__ __launch_bounds__(128) void vn_conv_fwd(const half_t* __restrict__ xhr,
                                                   const half_t* __restrict__ xhi,
                                                   const float* __restrict__ Dr,
                                                   const float* __restrict__ Di,
                                                   half_t* __restrict__ DXr,
                                                   half_t* __restrict__ DXi) {
  __shared__ __align__(16) half_t Bp[48][360];      // B^T [n][k], 34.6 KB
  __shared__ __align__(16) half_t P[2][KSZ][88];    // patch, core at col 8
  __shared__ __align__(16) half_t Ach[2][64][40];   // per-chunk A, m-major
  int tid = threadIdx.x;
  int blk = blockIdx.x;                // t*1024 + y*4 + xt
  int xt = blk & 3, y = (blk >> 2) & 255, t = blk >> 10;
  int x0 = xt * 64;

  __builtin_prefetch(Dr, 0, 1);
  __builtin_prefetch(Di, 0, 1);

  // zero patch (pads must be finite for WMMA)
  {
    unsigned* pz = (unsigned*)&P[0][0][0];
    for (int i = tid; i < (int)(sizeof(P) / 4); i += 128) pz[i] = 0u;
  }
  // build B^T: n<24 -> DXr col, n>=24 -> DXi col; zero on vp>=11 / k>=352
  for (int n = 0; n < 48; ++n) {
    int f = n % 24, g = n / 24;
    for (int k = tid; k < 360; k += 128) {
      int kb = k >> 4, vp = k & 15;
      int valid = (vp < 11) && (k < FKP);
      int c = kb >= 11;
      int u = kb - (c ? 11 : 0);
      if (u > 10) u = 10;
      int tap = u * 11 + (vp < 11 ? vp : 0);
      float w;
      if (g == 0) w = (c == 0) ?  Dr[f * TAPS + tap] : -Di[f * TAPS + tap];
      else        w = (c == 0) ?  Di[f * TAPS + tap] :  Dr[f * TAPS + tap];
      Bp[n][k] = valid ? (half_t)w : (half_t)0.f;
    }
  }
  __syncthreads();   // zero-fill visible before staging overwrites core/halo

  // stage patch: async 16B core segments + scalar halos (cols js+3, core js=5..68)
  for (int e = tid; e < 2 * KSZ * 8; e += 128) {
    int c = e / (KSZ * 8), r = (e / 8) % KSZ, seg = e & 7;
    int ys = y + r - 5;
    if (ys >= 0 && ys < HH)
      copy16_g2l((c ? xhi : xhr) + (t * HH + ys) * WW + x0 + seg * 8,
                 &P[c][r][8 + seg * 8]);
  }
  for (int e = tid; e < 2 * KSZ * 10; e += 128) {
    int c = e / (KSZ * 10), r = (e / 10) % KSZ, s = e % 10;
    int js = (s < 5) ? s : s + 64;
    int ys = y + r - 5, xs = x0 + js - 5;
    if (ys >= 0 && ys < HH && xs >= 0 && xs < WW)
      P[c][r][js + 3] = (c ? xhi : xhr)[(t * HH + ys) * WW + xs];
  }
  wait_g2l();
  __syncthreads();

  int wave = tid >> 5, lane = tid & 31, mrow = lane & 15, kh = lane >> 4;
  int mI = tid >> 1, bI = tid & 1;     // staging role: row mI, K-block bI
  v8f acc0 = {}, acc1 = {}, acc2 = {};
  for (int kc = 0; kc < 11; ++kc) {
    int buf = kc & 1;
    // stage 64x32 A chunk: contiguous 16-half copy per thread
    {
      int kb = kc * 2 + bI;
      int cB = kb >= 11;
      int uB = kb - (cB ? 11 : 0);
      const half_t* prow = &P[cB][uB][mI + 3];
      half_t* arow = &Ach[buf][mI][bI * 16];
#pragma unroll
      for (int q = 0; q < 16; ++q) arow[q] = prow[q];
    }
    __syncthreads();
    v8h alo = *(const v8h*)&Ach[buf][wave * 16 + mrow][kh * 16];
    v8h ahi = *(const v8h*)&Ach[buf][wave * 16 + mrow][kh * 16 + 8];
    v16h a = __builtin_shufflevector(alo, ahi, 0,1,2,3,4,5,6,7,8,9,10,11,12,13,14,15);
    int kb0 = kc * 32 + kh * 16;
    v8h b0l = *(const v8h*)&Bp[mrow][kb0],      b0h = *(const v8h*)&Bp[mrow][kb0 + 8];
    v8h b1l = *(const v8h*)&Bp[16 + mrow][kb0], b1h = *(const v8h*)&Bp[16 + mrow][kb0 + 8];
    v8h b2l = *(const v8h*)&Bp[32 + mrow][kb0], b2h = *(const v8h*)&Bp[32 + mrow][kb0 + 8];
    v16h b0 = __builtin_shufflevector(b0l, b0h, 0,1,2,3,4,5,6,7,8,9,10,11,12,13,14,15);
    v16h b1 = __builtin_shufflevector(b1l, b1h, 0,1,2,3,4,5,6,7,8,9,10,11,12,13,14,15);
    v16h b2 = __builtin_shufflevector(b2l, b2h, 0,1,2,3,4,5,6,7,8,9,10,11,12,13,14,15);
    acc0 = __builtin_amdgcn_wmma_f32_16x16x32_f16(false, a, false, b0, (short)0, acc0, false, false);
    acc1 = __builtin_amdgcn_wmma_f32_16x16x32_f16(false, a, false, b1, (short)0, acc1, false, false);
    acc2 = __builtin_amdgcn_wmma_f32_16x16x32_f16(false, a, false, b2, (short)0, acc2, false, false);
  }
  // C layout: VGPR r -> M = kh*8 + r, N = mrow (+16/32 per tile)
  int xb = x0 + wave * 16;
#pragma unroll
  for (int r = 0; r < 8; ++r) {
    int px = xb + kh * 8 + r;
    DXr[((t * FF + mrow) * HH + y) * WW + px] = (half_t)acc0[r];
    int n1 = 16 + mrow;
    if (n1 < 24) DXr[((t * FF + n1)        * HH + y) * WW + px] = (half_t)acc1[r];
    else         DXi[((t * FF + (n1 - 24)) * HH + y) * WW + px] = (half_t)acc1[r];
    DXi[((t * FF + 8 + mrow) * HH + y) * WW + px] = (half_t)acc2[r];
  }
}

// -------------------------------------------- spline activation (VTV weight)
__global__ __launch_bounds__(256) void vn_act(half_t* __restrict__ DXr,
                                              half_t* __restrict__ DXi,
                                              const float* __restrict__ knots) {
  int idx = blockIdx.x * 256 + threadIdx.x;
  if (idx >= FF * HW2) return;
  int f = idx >> 16, p = idx & 65535;
  float s = 0.f;
#pragma unroll
  for (int t = 0; t < TT; ++t) {
    float a = (float)DXr[(t * FF + f) * HW2 + p];
    float b = (float)DXi[(t * FF + f) * HW2 + p];
    s += a * a + b * b;
  }
  float act_in = sqrtf(s) * (1.f / 6.f);
  float pos = act_in * (float)(KN - 1);
  pos = fminf(fmaxf(pos, 0.f), (float)(KN - 1));
  int i0 = (int)pos; if (i0 > KN - 2) i0 = KN - 2;
  float fr = pos - (float)i0;
  float k0 = knots[i0 * FF + f], k1 = knots[(i0 + 1) * FF + f];
  float act = k0 * (1.f - fr) + k1 * fr;
#pragma unroll
  for (int t = 0; t < TT; ++t) {
    int o = (t * FF + f) * HW2 + p;
    DXr[o] = (half_t)((float)DXr[o] * act);
    DXi[o] = (half_t)((float)DXi[o] * act);
  }
}

// ------------------------------------------------ adjoint complex conv: WMMA
// Two phases (c = Vr/Vi). Per phase K=(f*11+u)*16+vp -> 4224 = 132 chunks.
// M=32 px/WG (2 waves). N-tile cols {rg_r, rg_i} (B zero elsewhere).
#define AKP 4224
__global__ __launch_bounds__(64) void vn_conv_adj(const half_t* __restrict__ Vr,
                                                  const half_t* __restrict__ Vi,
                                                  const float* __restrict__ Dr,
                                                  const float* __restrict__ Di,
                                                  float* __restrict__ rgr,
                                                  float* __restrict__ rgi) {
  __shared__ __align__(16) half_t Bt[2][4232];      // B^T per phase, 16.9 KB
  __shared__ __align__(16) half_t P[FF][KSZ][56];   // 24-plane patch, 29.6 KB
  __shared__ __align__(16) half_t Ach[2][32][40];   // per-chunk A, 5 KB
  int tid = threadIdx.x;
  int blk = blockIdx.x;                // t*2048 + y*8 + xt
  int xt = blk & 7, y = (blk >> 3) & 255, t = blk >> 11;
  int x0 = xt * 32;

  __builtin_prefetch(Dr, 0, 1);
  __builtin_prefetch(Di, 0, 1);

  {
    unsigned* pz = (unsigned*)&P[0][0][0];
    for (int i = tid; i < (int)(sizeof(P) / 4); i += 64) pz[i] = 0u;
  }

  int wave = tid >> 5, lane = tid & 31, mrow = lane & 15, kh = lane >> 4;
  int mI = tid >> 1, bI = tid & 1;
  v8f acc = {};
  for (int c = 0; c < 2; ++c) {
    __syncthreads();  // previous compute done; zero-fill visible on first pass
    // B^T (flip + transpose): col0 -> rg_r, col1 -> rg_i
    for (int n = 0; n < 2; ++n) {
      for (int k = tid; k < 4232; k += 64) {
        int kb = k >> 4, vp = k & 15;
        int valid = (vp < 11) && (k < AKP);
        int f = kb / 11; int u = kb - f * 11;
        if (f > 23) f = 23;
        if (u > 10) u = 10;
        int vpc = vp < 11 ? vp : 10;
        int flip = (10 - u) * KSZ + (10 - vpc);
        float drt = Dr[f * TAPS + flip], dit = Di[f * TAPS + flip];
        float w = (c == 0) ? (n == 0 ? drt : -dit) : (n == 0 ? dit : drt);
        Bt[n][k] = valid ? (half_t)w : (half_t)0.f;
      }
    }
    // stage 24-plane patch from V_c: async cores + scalar halos
    const half_t* src = c ? Vi : Vr;
    for (int e = tid; e < FF * KSZ * 4; e += 64) {
      int f = e / (KSZ * 4), r = (e / 4) % KSZ, seg = e & 3;
      int ys = y + r - 5;
      if (ys >= 0 && ys < HH)
        copy16_g2l(src + ((t * FF + f) * HH + ys) * WW + x0 + seg * 8,
                   &P[f][r][8 + seg * 8]);
    }
    for (int e = tid; e < FF * KSZ * 10; e += 64) {
      int f = e / (KSZ * 10), r = (e / 10) % KSZ, s = e % 10;
      int js = (s < 5) ? s : s + 32;
      int ys = y + r - 5, xs = x0 + js - 5;
      if (ys >= 0 && ys < HH && xs >= 0 && xs < WW)
        P[f][r][js + 3] = src[((t * FF + f) * HH + ys) * WW + xs];
    }
    wait_g2l();
    __syncthreads();

    for (int kc = 0; kc < 132; ++kc) {
      int buf = kc & 1;
      {
        int kb = kc * 2 + bI;
        int fB = kb / 11;
        int uB = kb - fB * 11;
        const half_t* prow = &P[fB][uB][mI + 3];
        half_t* arow = &Ach[buf][mI][bI * 16];
#pragma unroll
        for (int q = 0; q < 16; ++q) arow[q] = prow[q];
      }
      __syncthreads();
      v8h alo = *(const v8h*)&Ach[buf][wave * 16 + mrow][kh * 16];
      v8h ahi = *(const v8h*)&Ach[buf][wave * 16 + mrow][kh * 16 + 8];
      v16h a = __builtin_shufflevector(alo, ahi, 0,1,2,3,4,5,6,7,8,9,10,11,12,13,14,15);
      v8h z = {};
      v8h blo = z, bhi = z;
      if (mrow < 2) {
        int kb0 = kc * 32 + kh * 16;
        blo = *(const v8h*)&Bt[mrow][kb0];
        bhi = *(const v8h*)&Bt[mrow][kb0 + 8];
      }
      v16h b = __builtin_shufflevector(blo, bhi, 0,1,2,3,4,5,6,7,8,9,10,11,12,13,14,15);
      acc = __builtin_amdgcn_wmma_f32_16x16x32_f16(false, a, false, b,
                                                   (short)0, acc, false, false);
    }
  }
  if (mrow < 2) {
    float* dst = (mrow == 0) ? rgr : rgi;
    int xb = x0 + wave * 16;
#pragma unroll
    for (int r = 0; r < 8; ++r) {
      int px = xb + kh * 8 + r;
      dst[(t * HH + y) * WW + px] = acc[r];
    }
  }
}

// ---------------------------------------------- data-consistency mask subtract
__global__ __launch_bounds__(256) void vn_mask_sub(float* __restrict__ tr,
                                                   float* __restrict__ ti,
                                                   const float* __restrict__ mask,
                                                   const float* __restrict__ ksr,
                                                   const float* __restrict__ ksi) {
  int i = blockIdx.x * 256 + threadIdx.x;
  if (i >= NPIX) return;
  int t = i >> 16, w = i & 255;
  float mk = mask[t * WW + w];
  tr[i] = tr[i] * mk - ksr[i];
  ti[i] = ti[i] * mk - ksi[i];
}

// ------------------------------------------------------------ momentum update
__global__ __launch_bounds__(256) void vn_update(float* __restrict__ Xr, float* __restrict__ Xi,
                                                 float* __restrict__ mr_, float* __restrict__ mi_,
                                                 const float* __restrict__ dgr, const float* __restrict__ dgi,
                                                 const float* __restrict__ rgr, const float* __restrict__ rgi,
                                                 const float* __restrict__ alphas,
                                                 const float* __restrict__ moms, int l,
                                                 half_t* __restrict__ Xhr, half_t* __restrict__ Xhi) {
  int i = blockIdx.x * 256 + threadIdx.x;
  if (i >= NPIX) return;
  float a = alphas[l], mo = moms[l];
  float gr = a * dgr[i] + rgr[i];
  float gi = a * dgi[i] + rgi[i];
  float nr = mo * mr_[i] + gr;
  float ni = mo * mi_[i] + gi;
  mr_[i] = nr; mi_[i] = ni;
  float xr_ = Xr[i] - nr, xi_ = Xi[i] - ni;
  Xr[i] = xr_; Xi[i] = xi_;
  Xhr[i] = (half_t)xr_; Xhi[i] = (half_t)xi_;
}

// -------------------------------------------------------- denormalize + pack
__global__ __launch_bounds__(256) void vn_pack(const float* __restrict__ Xr,
                                               const float* __restrict__ Xi,
                                               const float* __restrict__ scal,
                                               float* __restrict__ out) {
  int i = blockIdx.x * 256 + threadIdx.x;
  if (i >= NPIX) return;
  float nm = scal[2];
  out[2 * i + 0] = Xr[i] * nm;
  out[2 * i + 1] = Xi[i] * nm;
}

// ============================================================================
extern "C" void kernel_launch(void* const* d_in, const int* in_sizes, int n_in,
                              void* d_out, int out_size, void* d_ws, size_t ws_size,
                              hipStream_t stream) {
  (void)in_sizes; (void)n_in; (void)out_size; (void)ws_size;

  const float* Xr_in = (const float*)d_in[0];
  const float* Xi_in = (const float*)d_in[1];
  const float* mask  = (const float*)d_in[2];
  const float* kr    = (const float*)d_in[3];   // (L,24,1,11,11)
  const float* ki    = (const float*)d_in[4];
  const float* kn    = (const float*)d_in[5];   // (L,35,24)
  const float* al    = (const float*)d_in[6];
  const float* mo    = (const float*)d_in[7];
  const float* k0    = (const float*)d_in[8];

  float* ws   = (float*)d_ws;
  float* scal = ws;                     // [0]=num [1]=den [2]=norm
  float* pnum = ws + 16;                // 256 partials
  float* pden = ws + 272;               // 256 partials
  float* arr  = ws + 1024;
  float* Xks_r  = arr + 0  * (size_t)NPIX;
  float* Xks_i  = arr + 1  * (size_t)NPIX;
  float* Ximg_r = arr + 2  * (size_t)NPIX;
  float* Ximg_i = arr + 3  * (size_t)NPIX;
  float* m_r    = arr + 4  * (size_t)NPIX;
  float* m_i    = arr + 5  * (size_t)NPIX;
  float* tmp_r  = arr + 6  * (size_t)NPIX;
  float* tmp_i  = arr + 7  * (size_t)NPIX;
  float* t2_r   = arr + 8  * (size_t)NPIX;
  float* t2_i   = arr + 9  * (size_t)NPIX;
  float* rg_r   = arr + 10 * (size_t)NPIX;
  float* rg_i   = arr + 11 * (size_t)NPIX;
  half_t* DXr = (half_t*)(arr + 12 * (size_t)NPIX);
  half_t* DXi = DXr + (size_t)TT * FF * HW2;
  half_t* Xh_r = DXi + (size_t)TT * FF * HW2;   // f16 mirror of X_img
  half_t* Xh_i = Xh_r + (size_t)NPIX;

  const int PB = (NPIX + 255) / 256;        // 1536 pointwise blocks
  const int FFTB = TT * 256;                // 1536 FFT lines per pass

  // normalize_ks_
  vn_reduce  <<<256, 256, 0, stream>>>(Xr_in, Xi_in, mask, pnum, pden);
  vn_finalize<<<1, 1, 0, stream>>>(pnum, pden, scal);
  vn_prep    <<<PB, 256, 0, stream>>>(Xr_in, Xi_in, scal, Xks_r, Xks_i,
                                      tmp_r, tmp_i, m_r, m_i);
  // X_img = k2i(X_ks) * k0   (inverse FFT: sign = +1)
  vn_fft256<<<FFTB, 256, 0, stream>>>(tmp_r, tmp_i, t2_r, t2_i, 0, +1.f);
  vn_fft256<<<FFTB, 256, 0, stream>>>(t2_r, t2_i, tmp_r, tmp_i, 1, +1.f);
  vn_scale_k0<<<PB, 256, 0, stream>>>(tmp_r, tmp_i, k0, Ximg_r, Ximg_i, Xh_r, Xh_i);

  for (int l = 0; l < 8; ++l) {
    const float* Drl = kr + (size_t)l * FF * TAPS;
    const float* Dil = ki + (size_t)l * FF * TAPS;
    const float* knl = kn + (size_t)l * KN * FF;
    // VTV regularizer: fwd conv (WMMA) -> spline activation -> adjoint (WMMA)
    vn_conv_fwd<<<TT * 256 * 4, 128, 0, stream>>>(Xh_r, Xh_i, Drl, Dil, DXr, DXi);
    vn_act     <<<(FF * HW2 + 255) / 256, 256, 0, stream>>>(DXr, DXi, knl);
    vn_conv_adj<<<TT * 256 * 8, 64, 0, stream>>>(DXr, DXi, Drl, Dil, rg_r, rg_i);
    // data consistency: k2i(i2k(X)*mask - X_ks)
    vn_fft256<<<FFTB, 256, 0, stream>>>(Ximg_r, Ximg_i, t2_r, t2_i, 0, -1.f);
    vn_fft256<<<FFTB, 256, 0, stream>>>(t2_r, t2_i, tmp_r, tmp_i, 1, -1.f);
    vn_mask_sub<<<PB, 256, 0, stream>>>(tmp_r, tmp_i, mask, Xks_r, Xks_i);
    vn_fft256<<<FFTB, 256, 0, stream>>>(tmp_r, tmp_i, t2_r, t2_i, 0, +1.f);
    vn_fft256<<<FFTB, 256, 0, stream>>>(t2_r, t2_i, tmp_r, tmp_i, 1, +1.f);
    // momentum step
    vn_update<<<PB, 256, 0, stream>>>(Ximg_r, Ximg_i, m_r, m_i,
                                      tmp_r, tmp_i, rg_r, rg_i, al, mo, l, Xh_r, Xh_i);
  }
  vn_pack<<<PB, 256, 0, stream>>>(Ximg_r, Ximg_i, scal, (float*)d_out);
}
